// AttentionBlock_51049981281080
// MI455X (gfx1250) — matrix operand order repaired and checked by reading, not compiled, not run on hardware
//
#include <hip/hip_runtime.h>
#include <hip/hip_bf16.h>
#include <math.h>

typedef __attribute__((ext_vector_type(16))) _Float16 v16h;
typedef __attribute__((ext_vector_type(8)))  _Float16 v8h;
typedef __attribute__((ext_vector_type(8)))  float    v8f;
typedef __attribute__((ext_vector_type(4)))  unsigned int v4u;
typedef __attribute__((ext_vector_type(8)))  int      v8i;
typedef __attribute__((ext_vector_type(4)))  int      v4i;

#define QS   0.21022410381342863f   /* 512^-0.25 */
#define EPS_ 1e-5f

__device__ __forceinline__ v8f wmma16(v16h a, v16h b, v8f c) {
  // D = A(16x32 f16) * B(32x16 f16) + C(16x16 f32)
  return __builtin_amdgcn_wmma_f32_16x16x32_f16(false, a, false, b, (short)0, c,
                                                false, false);
}

// A-fragment from an LDS tile stored row-major [16 rows M][K...] (f16).
__device__ __forceinline__ v16h ldsA(const _Float16* t, int stride, int lane) {
  int row = lane & 15;
  int kb  = (lane >> 4) << 4;
  const _Float16* p = t + row * stride + kb;
  v8h lo = *(const v8h*)p;
  v8h hi = *(const v8h*)(p + 8);
  return __builtin_shufflevector(lo, hi, 0,1,2,3,4,5,6,7,8,9,10,11,12,13,14,15);
}

// B-fragment from an LDS tile stored [16 rows N][K...] (f16).
__device__ __forceinline__ v16h ldsB(const _Float16* t, int stride, int lane) {
  int col = lane & 15;
  int kh  = (lane >> 4) << 4;
  return *(const v16h*)(t + col * stride + kh);
}

// Tensor Data Mover: async 2-D tile load (f16 elements) global -> LDS.
// D# built per CDNA5 ISA (group0 §8.3, group1 §8.4).
__device__ __forceinline__ void tdm_load_2d_f16(
    unsigned lds_off, const _Float16* gptr,
    unsigned tensor_d0, unsigned tensor_d1,
    unsigned tile_d0, unsigned tile_d1, unsigned stride0) {
  unsigned long long ga = (unsigned long long)(uintptr_t)gptr;
  v4u g0;
  g0[0] = 1u;                                        // count=1, user D#
  g0[1] = lds_off;                                   // lds_addr (bytes)
  g0[2] = (unsigned)(ga & 0xffffffffu);              // global_addr[31:0]
  g0[3] = (unsigned)((ga >> 32) & 0x1ffffffu)        // global_addr[56:32]
        | (2u << 30);                                // type=2 ("image")
  v8i g1;
  g1[0] = (int)(1u << 16);                           // data_size=1 (2 bytes)
  g1[1] = (int)((tensor_d0 & 0xffffu) << 16);        // tensor_dim0[15:0]
  g1[2] = (int)((tensor_d0 >> 16) | ((tensor_d1 & 0xffffu) << 16));
  g1[3] = (int)((tensor_d1 >> 16) | (tile_d0 << 16));// tile_dim0
  g1[4] = (int)(tile_d1 & 0xffffu);                  // tile_dim1 (tile_dim2=0)
  g1[5] = (int)stride0;                              // tensor_dim0_stride[31:0]
  g1[6] = 0;
  g1[7] = 0;
  v4i z4 = {0, 0, 0, 0};
#if __clang_major__ >= 23
  v8i z8 = {0, 0, 0, 0, 0, 0, 0, 0};
  __builtin_amdgcn_tensor_load_to_lds(g0, g1, z4, z4, z8, 0);
#else
  __builtin_amdgcn_tensor_load_to_lds(g0, g1, z4, z4, 0);
#endif
}

// ---------------------------------------------------------------------------
// Kernel 1: per (batch, group) mean / rstd.  1024 blocks x 256 threads.
// ---------------------------------------------------------------------------
__global__ __launch_bounds__(256) void gn_stats_kernel(
    const float* __restrict__ x, float* __restrict__ stats) {
  int bg = blockIdx.x;
  const float4* p = (const float4*)(x + (size_t)bg * 16384);
  int t = threadIdx.x;
  float s1 = 0.f, s2 = 0.f;
#pragma unroll
  for (int i = 0; i < 16; ++i) {
    float4 v = p[t + i * 256];
    s1 += v.x + v.y + v.z + v.w;
    s2 += v.x*v.x + v.y*v.y + v.z*v.z + v.w*v.w;
  }
  __shared__ float r1[256], r2[256];
  r1[t] = s1; r2[t] = s2;
  __syncthreads();
  for (int s = 128; s > 0; s >>= 1) {
    if (t < s) { r1[t] += r1[t+s]; r2[t] += r2[t+s]; }
    __syncthreads();
  }
  if (t == 0) {
    float mean = r1[0] * (1.f/16384.f);
    float var  = r2[0] * (1.f/16384.f) - mean*mean;
    stats[bg*2]   = mean;
    stats[bg*2+1] = rsqrtf(var + EPS_);
  }
}

// ---------------------------------------------------------------------------
// Kernel 2: qkv[b] = qkv_w * groupnorm(x[b]) + qkv_b  (f16 out, q/k pre-scaled
// by 512^-0.25 folded into W and bias rows).  128(O) x 128(P) tile per block.
// ---------------------------------------------------------------------------
__global__ __launch_bounds__(256) void qkv_kernel(
    const float* __restrict__ x, const float* __restrict__ stats,
    const float* __restrict__ gnw, const float* __restrict__ gnb,
    const float* __restrict__ w, const float* __restrict__ bias,
    _Float16* __restrict__ qkvh) {
  int b  = blockIdx.z;
  int o0 = blockIdx.y * 128;
  int p0 = blockIdx.x * 128;
  int t = threadIdx.x, lane = t & 31, wv = t >> 5;
  __shared__ _Float16 sA[128 * 32];   // weights  [o][c]
  __shared__ _Float16 sB[128 * 32];   // norm act [p][c] (transposed)
  int wr = wv >> 2, wc = wv & 3;
  v8f acc[4][2] = {};

  for (int kc = 0; kc < 16; ++kc) {
    int c0 = kc * 32;
    __syncthreads();
    {   // stage weight tile (fold q/k scale)
      int r = t >> 1, h = (t & 1) * 16;
      int o = o0 + r;
      float scl = (o < 1024) ? QS : 1.0f;
      const float* src = w + o * 512 + c0 + h;
      _Float16* dst = sA + r * 32 + h;
#pragma unroll
      for (int i = 0; i < 16; ++i) dst[i] = (_Float16)(src[i] * scl);
    }
    {   // stage normalized-x tile, transposed to [p][c]
      int cl = t >> 3, pch = (t & 7) * 16;
      int c = c0 + cl;
      float mean = stats[(b * 32 + (c >> 4)) * 2];
      float rstd = stats[(b * 32 + (c >> 4)) * 2 + 1];
      float ww = gnw[c] * rstd;
      float bb = gnb[c] - mean * ww;
      const float* src = x + ((size_t)(b * 512 + c) << 10) + p0 + pch;
#pragma unroll
      for (int i = 0; i < 16; ++i)
        sB[(pch + i) * 32 + cl] = (_Float16)(src[i] * ww + bb);
    }
    __syncthreads();
    v16h af[4], bf[2];
#pragma unroll
    for (int mi = 0; mi < 4; ++mi) af[mi] = ldsA(sA + (wr*64 + mi*16)*32, 32, lane);
#pragma unroll
    for (int ni = 0; ni < 2; ++ni) bf[ni] = ldsB(sB + (wc*32 + ni*16)*32, 32, lane);
#pragma unroll
    for (int mi = 0; mi < 4; ++mi)
#pragma unroll
      for (int ni = 0; ni < 2; ++ni)
        acc[mi][ni] = wmma16(af[mi], bf[ni], acc[mi][ni]);
  }
  int mrow = (lane >> 4) << 3, ncol = lane & 15;
#pragma unroll
  for (int mi = 0; mi < 4; ++mi)
#pragma unroll
    for (int ni = 0; ni < 2; ++ni)
#pragma unroll
      for (int r = 0; r < 8; ++r) {
        int o = o0 + wr*64 + mi*16 + r + mrow;
        int p = p0 + wc*32 + ni*16 + ncol;
        float scl = (o < 1024) ? QS : 1.0f;
        qkvh[((b*1536 + o) << 10) + p] = (_Float16)(acc[mi][ni][r] + bias[o]*scl);
      }
}

// ---------------------------------------------------------------------------
// Kernel 3: fused attention + proj + residual.  One block = (batch, 32-query
// tile), 8 waves, 192KB dynamic LDS.  v tiles staged by the TDM
// (tensor_load_to_lds) double-buffered against WMMA compute.
// ---------------------------------------------------------------------------
__global__ __launch_bounds__(256) void attn_kernel(
    const float* __restrict__ x, const _Float16* __restrict__ qkvh,
    const float* __restrict__ pw, const float* __restrict__ pb,
    float* __restrict__ out) {
  extern __shared__ char smem[];
  float*    S   = (float*)smem;                  // [32][1024] f32 (phase 1-2)
  _Float16* sQ  = (_Float16*)(smem + 131072);    // [32][512]  f16 (phase 1)
  _Float16* sK  = (_Float16*)(smem + 163840);    // [256][32]  f16 (phase 1)
  _Float16* sP  = (_Float16*)(smem + 131072);    // [32][1024] f16 (phase 2-3)
  _Float16* sV0 = (_Float16*)smem;               // [512][32]  f16 (phase 3, buf0)
  _Float16* sV1 = (_Float16*)(smem + 32768);     // [512][32]  f16 (phase 3, buf1)
  _Float16* sW  = (_Float16*)(smem + 65536);     // [512][32]  f16 (phase 4)
  _Float16* sH  = (_Float16*)(smem + 98304);     // [32][512]  f16 (phase 3-4)

  int b  = blockIdx.y;
  int p0 = blockIdx.x * 32;
  int t = threadIdx.x, lane = t & 31, wv = t >> 5;
  int qbase = b * 1536;
  int m  = wv & 1;       // query sub-tile (16 rows)
  int wp = wv >> 1;      // 0..3 column group

  // ---- load this tile's q columns, transposed to sQ[p][c]
#pragma unroll
  for (int rr = 0; rr < 2; ++rr) {
    int c = (t << 1) + rr;
    const v8h* qrow = (const v8h*)(qkvh + ((qbase + c) << 10) + p0);
#pragma unroll
    for (int seg = 0; seg < 4; ++seg) {
      v8h v = qrow[seg];
#pragma unroll
      for (int j = 0; j < 8; ++j) sQ[(seg*8 + j) * 512 + c] = v[j];
    }
  }

  // ---- phase 1: S[32][1024] = q^T k  (scale already folded in)
  for (int jt = 0; jt < 4; ++jt) {
    int j0 = jt * 256;
    v8f acc[4] = {};
    for (int kc = 0; kc < 16; ++kc) {
      int c0 = kc * 32;
      __syncthreads();
      {   // stage k chunk transposed to sK[j][c]
        int cl = t >> 3, jch = (t & 7) * 32;
        const v8h* kr = (const v8h*)(qkvh + ((qbase + 512 + c0 + cl) << 10) + j0 + jch);
#pragma unroll
        for (int seg = 0; seg < 4; ++seg) {
          v8h v = kr[seg];
#pragma unroll
          for (int j = 0; j < 8; ++j) sK[(jch + seg*8 + j) * 32 + cl] = v[j];
        }
      }
      __syncthreads();
      v16h a = ldsA(sQ + (m*16)*512 + c0, 512, lane);
#pragma unroll
      for (int i = 0; i < 4; ++i) {
        v16h bf = ldsB(sK + ((wp + 4*i) * 16) * 32, 32, lane);
        acc[i] = wmma16(a, bf, acc[i]);
      }
    }
    int mrow = (lane >> 4) << 3, ncol = lane & 15;
#pragma unroll
    for (int i = 0; i < 4; ++i)
#pragma unroll
      for (int r = 0; r < 8; ++r)
        S[(m*16 + r + mrow) * 1024 + j0 + (wp + 4*i)*16 + ncol] = acc[i][r];
  }
  __syncthreads();

  // ---- phase 2: row softmax, emit P (f16)
  for (int rr = 0; rr < 4; ++rr) {
    int row = wv * 4 + rr;
    float vals[32];
    float mx = -1e30f;
#pragma unroll
    for (int i = 0; i < 32; ++i) {
      vals[i] = S[row*1024 + lane + 32*i];
      mx = fmaxf(mx, vals[i]);
    }
#pragma unroll
    for (int off = 16; off > 0; off >>= 1) mx = fmaxf(mx, __shfl_xor(mx, off, 32));
    float sm = 0.f;
#pragma unroll
    for (int i = 0; i < 32; ++i) { vals[i] = __expf(vals[i] - mx); sm += vals[i]; }
#pragma unroll
    for (int off = 16; off > 0; off >>= 1) sm += __shfl_xor(sm, off, 32);
    float inv = 1.0f / sm;
#pragma unroll
    for (int i = 0; i < 32; ++i)
      sP[row*1024 + lane + 32*i] = (_Float16)(vals[i] * inv);
  }
  __syncthreads();   // S dead from here; sV buffers may be written

  // ---- phase 3: hT[32][512] = P (32x1024) * v^T (1024x512)
  // v tiles [512 c][32 j] are pure 2-D copies -> Tensor Data Mover,
  // double-buffered: TDM transfer of chunk kj+1 overlaps WMMA of chunk kj.
  {
    const _Float16* vbase = qkvh + ((qbase + 1024) << 10);
    unsigned lds_v0 = (unsigned)(uintptr_t)(void*)sV0;
    unsigned lds_v1 = (unsigned)(uintptr_t)(void*)sV1;
    if (wv == 0)   // prologue: fetch chunk 0
      tdm_load_2d_f16(lds_v0, vbase, 1024u, 512u, 32u, 512u, 1024u);

    v8f acc[8] = {};
    for (int kj = 0; kj < 32; ++kj) {
      int j0 = kj * 32;
      __syncthreads();   // everyone done reading the buffer TDM will overwrite
      if (wv == 0) {
        if (kj < 31) {
          tdm_load_2d_f16(((kj + 1) & 1) ? lds_v1 : lds_v0,
                          vbase + (j0 + 32), 1024u, 512u, 32u, 512u, 1024u);
          __builtin_amdgcn_s_wait_tensorcnt(1);  // chunk kj landed
        } else {
          __builtin_amdgcn_s_wait_tensorcnt(0);  // last chunk landed
        }
      }
      __syncthreads();   // publish chunk kj to all waves
      const _Float16* sV = (kj & 1) ? sV1 : sV0;
      v16h a = ldsA(sP + (m*16)*1024 + j0, 1024, lane);
#pragma unroll
      for (int i = 0; i < 8; ++i) {
        v16h bf = ldsB(sV + ((wp + 4*i) * 16) * 32, 32, lane);
        acc[i] = wmma16(a, bf, acc[i]);
      }
    }
    int mrow = (lane >> 4) << 3, ncol = lane & 15;
#pragma unroll
    for (int i = 0; i < 8; ++i)
#pragma unroll
      for (int r = 0; r < 8; ++r)
        sH[(m*16 + r + mrow) * 512 + (wp + 4*i)*16 + ncol] = (_Float16)acc[i][r];
  }
  __syncthreads();

  // ---- phase 4: out = x + proj_w * h + proj_b
  {
    int n = wv & 1;
    v8f acc[8] = {};
    for (int kc = 0; kc < 16; ++kc) {
      int c0 = kc * 32;
      __syncthreads();
#pragma unroll
      for (int rr = 0; rr < 2; ++rr) {           // stage proj_w chunk f32->f16
        int o = (t << 1) + rr;
        const float* src = pw + o * 512 + c0;
        _Float16* dst = sW + o * 32;
#pragma unroll
        for (int i = 0; i < 32; ++i) dst[i] = (_Float16)src[i];
      }
      __syncthreads();
      v16h bf = ldsB(sH + (n*16)*512 + c0, 512, lane);
#pragma unroll
      for (int i = 0; i < 8; ++i) {
        v16h a = ldsA(sW + ((wp + 4*i) * 16) * 32, 32, lane);
        acc[i] = wmma16(a, bf, acc[i]);
      }
    }
    int mrow = (lane >> 4) << 3, ncol = lane & 15;
#pragma unroll
    for (int i = 0; i < 8; ++i)
#pragma unroll
      for (int r = 0; r < 8; ++r) {
        int o = (wp + 4*i)*16 + r + mrow;
        int p = p0 + n*16 + ncol;
        int idx = ((b*512 + o) << 10) + p;
        out[idx] = x[idx] + pb[o] + acc[i][r];
      }
  }
}

// ---------------------------------------------------------------------------
extern "C" void kernel_launch(void* const* d_in, const int* in_sizes, int n_in,
                              void* d_out, int out_size, void* d_ws, size_t ws_size,
                              hipStream_t stream) {
  const float* x    = (const float*)d_in[0];
  const float* gnw  = (const float*)d_in[1];
  const float* gnb  = (const float*)d_in[2];
  const float* qkvw = (const float*)d_in[3];
  const float* qkvb = (const float*)d_in[4];
  const float* pw   = (const float*)d_in[5];
  const float* pb   = (const float*)d_in[6];
  float* out = (float*)d_out;

  float*    stats = (float*)d_ws;                              // 8 KB
  _Float16* qkvh  = (_Float16*)((char*)d_ws + 8192);           // 96 MB f16 q,k,v

  (void)in_sizes; (void)n_in; (void)out_size; (void)ws_size;

  hipFuncSetAttribute(reinterpret_cast<const void*>(attn_kernel),
                      hipFuncAttributeMaxDynamicSharedMemorySize, 196608);

  gn_stats_kernel<<<1024, 256, 0, stream>>>(x, stats);
  qkv_kernel<<<dim3(8, 12, 32), 256, 0, stream>>>(x, stats, gnw, gnb, qkvw, qkvb, qkvh);
  attn_kernel<<<dim3(32, 32), 256, 196608, stream>>>(x, qkvh, pw, pb, out);
}
